// GCN_9844065042433
// MI455X (gfx1250) — matrix operand order repaired and checked by reading, not compiled, not run on hardware
//
#include <hip/hip_runtime.h>

// ---------- types for WMMA ----------
typedef __attribute__((ext_vector_type(16))) __bf16 v16bf;
typedef __attribute__((ext_vector_type(8)))  float  v8f;

union BF16x16 { unsigned u[8]; uint4 q[2]; v16bf v; };

// Pack two f32 into one dword of two bf16 (truncating) with a single v_perm_b32.
// v_perm treats {S0,S1} as bytes 7..4 / 3..0; sel 0x07060302 picks
// [lo.b2, lo.b3, hi.b2, hi.b3] -> low half = lo>>16, high half = hi>>16.
__device__ __forceinline__ unsigned pkbf(float lo, float hi) {
    return __builtin_amdgcn_perm(__float_as_uint(hi), __float_as_uint(lo), 0x07060302u);
}

// ---------- generic fill ----------
__global__ void fill_kernel(float* __restrict__ p, long n, float v) {
    long i = (long)blockIdx.x * blockDim.x + threadIdx.x;
    if (i < n) p[i] = v;
}

// ---------- degree accumulation (weights onto dst) ----------
__global__ void deg_kernel(const int* __restrict__ dst, const float* __restrict__ ew,
                           float* __restrict__ deg, int E) {
    int e = blockIdx.x * blockDim.x + threadIdx.x;
    if (e < E) atomicAdd(&deg[dst[e]], ew[e]);
}

// ---------- dinv = rsqrt(deg) in place (deg >= 1 always: self-loop) ----------
__global__ void dinv_kernel(float* __restrict__ deg, int Nn) {
    int i = blockIdx.x * blockDim.x + threadIdx.x;
    if (i < Nn) deg[i] = rsqrtf(deg[i]);
}

// ---------- pre-pack W1 [512,16] f32 -> per-lane bf16 B fragments ----------
// Layout: packedB[(kstep*32 + lane)*8 + j] ; lane holds column (lane&15),
// K rows (lane>=16 ? 16 : 0) + 2j, 2j+1 inside the 32-deep K chunk.
__global__ void packb_kernel(const float* __restrict__ W1, unsigned* __restrict__ packedB) {
    int t = blockIdx.x * blockDim.x + threadIdx.x;   // [0, 512)
    if (t >= 16 * 32) return;
    int kstep = t >> 5;
    int lane  = t & 31;
    int ncol  = lane & 15;
    int kbase = kstep * 32 + ((lane >> 4) << 4);     // + half*16
    unsigned* outp = packedB + (size_t)t * 8;
    #pragma unroll
    for (int j = 0; j < 8; ++j) {
        float lo = W1[(kbase + 2 * j)     * 16 + ncol];
        float hi = W1[(kbase + 2 * j + 1) * 16 + ncol];
        outp[j] = pkbf(lo, hi);
    }
}

// ---------- GEMM1: xw[N,16] = x[N,512] @ W1[512,16] via bf16 WMMA ----------
// One wave per 16-row tile; 4 waves per block. B fragments preconverted.
__global__ void gemm1_wmma(const float* __restrict__ x, const unsigned* __restrict__ packedB,
                           float* __restrict__ xw, int Nn) {
    const int lane = threadIdx.x & 31;
    const int wid  = threadIdx.x >> 5;
    const int tile = blockIdx.x * 4 + wid;
    const int row0 = tile * 16;
    if (row0 >= Nn) return;

    const int m    = lane & 15;
    const int half = lane >> 4;          // 0: lanes 0-15, 1: lanes 16-31
    const int row  = row0 + m;
    const float* xrow = x + (size_t)(row < Nn ? row : 0) * 512;

    const int kAoff = half * 16;         // A-frag K offset inside 32-chunk
    const int ncol  = m;                 // C/D column held by this lane
    const uint4* bfrag = (const uint4*)(packedB + (size_t)lane * 8);

    v8f acc = {};
    #pragma unroll 2
    for (int kb = 0; kb < 512; kb += 32) {
        __builtin_prefetch(xrow + kb + 64, 0, 1);   // speculative; OOB dropped

        // A fragment: two contiguous 8-float runs -> 4x global_load_b128,
        // packed to bf16 pairs with 8x v_perm_b32.
        const float4* p0 = (const float4*)(xrow + kb + kAoff);
        const float4* p1 = (const float4*)(xrow + kb + kAoff + 8);
        float4 q0 = p0[0], q1 = p0[1], q2 = p1[0], q3 = p1[1];
        BF16x16 A;
        A.u[0] = pkbf(q0.x, q0.y); A.u[1] = pkbf(q0.z, q0.w);
        A.u[2] = pkbf(q1.x, q1.y); A.u[3] = pkbf(q1.z, q1.w);
        A.u[4] = pkbf(q2.x, q2.y); A.u[5] = pkbf(q2.z, q2.w);
        A.u[6] = pkbf(q3.x, q3.y); A.u[7] = pkbf(q3.z, q3.w);

        // B fragment: 32 coalesced bytes per lane -> 2x global_load_b128
        BF16x16 B;
        const uint4* bp = bfrag + (size_t)(kb >> 5) * 64;  // 32 lanes * 2 uint4
        B.q[0] = bp[0];
        B.q[1] = bp[1];

        acc = __builtin_amdgcn_wmma_f32_16x16x32_bf16(
                  false, A.v, false, B.v, (short)0, acc, false, false);
    }

    if (row0 + 16 <= Nn) {
        // full tile: unguarded strided stores (common/only path for N%16==0)
        float* outp = xw + (size_t)(row0 + half * 8) * 16 + ncol;
        #pragma unroll
        for (int r = 0; r < 8; ++r) outp[(size_t)r * 16] = acc[r];
    } else {
        #pragma unroll
        for (int r = 0; r < 8; ++r) {
            int orow = row0 + half * 8 + r;
            if (orow < Nn) xw[(size_t)orow * 16 + ncol] = acc[r];
        }
    }
}

// ---------- layer-1 edge scatter: agg1[dst] += norm * xw[src] ----------
__global__ void agg1_kernel(const int* __restrict__ src, const int* __restrict__ dst,
                            const float* __restrict__ ew, const float* __restrict__ dinv,
                            const float* __restrict__ xw, float* __restrict__ agg, int E) {
    int e = blockIdx.x * blockDim.x + threadIdx.x;
    if (e >= E) return;
    int s = src[e], d = dst[e];
    float nr = dinv[s] * ew[e] * dinv[d];
    const float4* hs = (const float4*)(xw + (size_t)s * 16);
    float* ad = agg + (size_t)d * 16;
    #pragma unroll
    for (int q = 0; q < 4; ++q) {
        float4 v = hs[q];
        atomicAdd(ad + 4 * q + 0, v.x * nr);
        atomicAdd(ad + 4 * q + 1, v.y * nr);
        atomicAdd(ad + 4 * q + 2, v.z * nr);
        atomicAdd(ad + 4 * q + 3, v.w * nr);
    }
}

// ---------- finalize1 (in place): h1 = relu(agg1 + dinv^2 * xw + b1) ----------
__global__ void fin1_kernel(float* __restrict__ agg, const float* __restrict__ xw,
                            const float* __restrict__ dinv, const float* __restrict__ b1,
                            int Nn) {
    int t = blockIdx.x * blockDim.x + threadIdx.x;
    if (t >= Nn * 16) return;
    int nn = t >> 4, f = t & 15;
    float di = dinv[nn];
    float v = agg[t] + di * di * xw[t] + b1[f];
    agg[t] = v > 0.0f ? v : 0.0f;
}

// ---------- GEMM2: z[N,40] = h1[N,16] @ W2[16,40] (tiny; scalar FMA) ----------
__global__ void gemm2_kernel(const float* __restrict__ h1, const float* __restrict__ W2,
                             float* __restrict__ z, int Nn) {
    int t = blockIdx.x * blockDim.x + threadIdx.x;
    if (t >= Nn * 40) return;
    int nn = t / 40, c = t % 40;
    const float* hr = h1 + (size_t)nn * 16;
    float s = 0.0f;
    #pragma unroll
    for (int k = 0; k < 16; ++k) s += hr[k] * W2[k * 40 + c];
    z[t] = s;
}

// ---------- layer-2 edge scatter ----------
__global__ void agg2_kernel(const int* __restrict__ src, const int* __restrict__ dst,
                            const float* __restrict__ ew, const float* __restrict__ dinv,
                            const float* __restrict__ z, float* __restrict__ agg, int E) {
    int e = blockIdx.x * blockDim.x + threadIdx.x;
    if (e >= E) return;
    int s = src[e], d = dst[e];
    float nr = dinv[s] * ew[e] * dinv[d];
    const float* zs = z + (size_t)s * 40;
    float* ad = agg + (size_t)d * 40;
    #pragma unroll
    for (int q = 0; q < 40; ++q) atomicAdd(ad + q, zs[q] * nr);
}

// ---------- finalize2 + log_softmax ----------
__global__ void fin2_kernel(const float* __restrict__ agg2, const float* __restrict__ z,
                            const float* __restrict__ dinv, const float* __restrict__ b2,
                            float* __restrict__ out, int Nn) {
    int nn = blockIdx.x * blockDim.x + threadIdx.x;
    if (nn >= Nn) return;
    float di = dinv[nn], d2 = di * di;
    const float* ar = agg2 + (size_t)nn * 40;
    const float* zr = z + (size_t)nn * 40;
    float v[40];
    float mx = -__builtin_inff();
    #pragma unroll
    for (int c = 0; c < 40; ++c) {
        float t = ar[c] + d2 * zr[c] + b2[c];
        v[c] = t;
        mx = fmaxf(mx, t);
    }
    float se = 0.0f;
    #pragma unroll
    for (int c = 0; c < 40; ++c) se += __expf(v[c] - mx);
    float lse = mx + __logf(se);
    float* orow = out + (size_t)nn * 40;
    #pragma unroll
    for (int c = 0; c < 40; ++c) orow[c] = v[c] - lse;
}

extern "C" void kernel_launch(void* const* d_in, const int* in_sizes, int n_in,
                              void* d_out, int out_size, void* d_ws, size_t ws_size,
                              hipStream_t stream) {
    const float* x  = (const float*)d_in[0];   // [N,512]
    const int*   ei = (const int*)  d_in[1];   // [2,E]
    const float* ew = (const float*)d_in[2];   // [E]
    const float* W1 = (const float*)d_in[3];   // [512,16]
    const float* b1 = (const float*)d_in[4];   // [16]
    const float* W2 = (const float*)d_in[5];   // [16,40]
    const float* b2 = (const float*)d_in[6];   // [40]
    float* out = (float*)d_out;

    const int F = 512;
    const int Nn = in_sizes[0] / F;
    const int E  = in_sizes[2];
    const int* src = ei;
    const int* dst = ei + E;

    // workspace carve-up (floats)
    float* ws   = (float*)d_ws;
    float* dinv = ws;                               // N
    float* xw   = dinv + (size_t)Nn;                // 16N
    float* agg1 = xw   + (size_t)Nn * 16;           // 16N (becomes h1)
    float* z    = agg1 + (size_t)Nn * 16;           // 40N
    float* agg2 = z    + (size_t)Nn * 40;           // 40N
    unsigned* packedB = (unsigned*)(agg2 + (size_t)Nn * 40);  // 4096 dwords (16 KB)

    const int B = 256;
    // 1) degrees: init with self-loop weight 1, add edge weights, invert-sqrt
    fill_kernel<<<(Nn + B - 1) / B, B, 0, stream>>>(dinv, (long)Nn, 1.0f);
    fill_kernel<<<(int)(((long)Nn * 16 + B - 1) / B), B, 0, stream>>>(agg1, (long)Nn * 16, 0.0f);
    fill_kernel<<<(int)(((long)Nn * 40 + B - 1) / B), B, 0, stream>>>(agg2, (long)Nn * 40, 0.0f);
    deg_kernel<<<(E + B - 1) / B, B, 0, stream>>>(dst, ew, dinv, E);
    dinv_kernel<<<(Nn + B - 1) / B, B, 0, stream>>>(dinv, Nn);

    // 2) xw = x @ W1 (WMMA bf16, f32 accumulate); B pre-packed to fragment layout
    packb_kernel<<<2, 256, 0, stream>>>(W1, packedB);
    int tiles = (Nn + 15) / 16;
    gemm1_wmma<<<(tiles + 3) / 4, 128, 0, stream>>>(x, packedB, xw, Nn);

    // 3) layer-1 propagation + finalize (relu, bias, self-loop)
    agg1_kernel<<<(E + B - 1) / B, B, 0, stream>>>(src, dst, ew, dinv, xw, agg1, E);
    fin1_kernel<<<(Nn * 16 + B - 1) / B, B, 0, stream>>>(agg1, xw, dinv, b1, Nn);

    // 4) z = h1 @ W2
    gemm2_kernel<<<(Nn * 40 + B - 1) / B, B, 0, stream>>>(agg1, W2, z, Nn);

    // 5) layer-2 propagation + finalize + log_softmax
    agg2_kernel<<<(E + B - 1) / B, B, 0, stream>>>(src, dst, ew, dinv, z, agg2, E);
    fin2_kernel<<<(Nn + B - 1) / B, B, 0, stream>>>(agg2, z, dinv, b2, out, Nn);
}
